// GCN_9191230013614
// MI455X (gfx1250) — matrix operand order repaired and checked by reading, not compiled, not run on hardware
//
#include <hip/hip_runtime.h>
#include <hip/hip_bf16.h>

// ---- problem constants (from reference) ----
#define NNODES   50000
#define NFEAT    512
#define NHID     256
#define NCLASS   40
#define NCLASS_P 48           // padded to 3x16 WMMA tiles
#define NEDGES   1600000

typedef __attribute__((ext_vector_type(16))) __bf16 v16bf;
typedef __attribute__((ext_vector_type(8)))  float  v8f;

__device__ __forceinline__ unsigned short f32_bf16(float f) {
    union { float f; unsigned u; } v; v.f = f;
    unsigned r = v.u + 0x7FFFu + ((v.u >> 16) & 1u);   // round-to-nearest-even
    return (unsigned short)(r >> 16);
}
__device__ __forceinline__ unsigned pack2(float a, float b) {
    return (unsigned)f32_bf16(a) | ((unsigned)f32_bf16(b) << 16);
}
__device__ __forceinline__ uint4 pack8(float4 a, float4 b) {
    uint4 p;
    p.x = pack2(a.x, a.y); p.y = pack2(a.z, a.w);
    p.z = pack2(b.x, b.y); p.w = pack2(b.z, b.w);
    return p;
}
__device__ __forceinline__ unsigned pcg_hash(unsigned x) {
    x = x * 747796405u + 2891336453u;
    x = ((x >> ((x >> 28) + 4u)) ^ x) * 277803737u;
    return (x >> 22) ^ x;
}

// fragment alias: 16 bf16 == two 128-bit LDS loads
union FragU { uint4 q[2]; v16bf bf; };

// =====================================================================
// GEMM1: H[50000x256] = X[50000x512] @ W1[512x256], bf16 WMMA, f32 accum
// block = 256 thr (8 waves as 4(M) x 2(N)); wave tile 16x32 (2 accum)
// block tile 64(M) x 64(N), K-step 32.
// LDS: A row-major [64][32], B transposed [n][k] -> all fragment loads
// are contiguous 128-bit ds_load_b128.
// =====================================================================
__launch_bounds__(256)
__global__ void gemm1_wmma(const float* __restrict__ X,
                           const float* __restrict__ W,
                           float* __restrict__ H) {
    __shared__ __align__(16) unsigned short As[64][32];   // 4 KB
    __shared__ __align__(16) unsigned short Bt[64][32];   // 4 KB (B transposed)
    const int tid  = threadIdx.x;
    const int lane = tid & 31;
    const int wid  = tid >> 5;
    const int wm   = wid & 3;          // 0..3 : M sub-tile
    const int wn   = wid >> 2;         // 0..1 : N sub-tile (each 32 wide)
    const int m0   = blockIdx.x * 64;
    const int n0   = blockIdx.y * 64;

    const int nlo = lane & 15;
    const int h   = lane >> 4;         // half-wave select
    const int mr  = wm * 16 + nlo;     // A fragment row
    const int kh  = 16 * h;            // A fragment K base
    const int kb  = 16 * h;            // B fragment K base

    v8f acc0 = {}, acc1 = {};
    for (int k0 = 0; k0 < NFEAT; k0 += 32) {
        { // stage A 64x32: 8 floats/thread -> one ds_store_b128
            const int row = tid >> 2;
            const int col = (tid & 3) * 8;
            const int g   = m0 + row;
            float4 v0 = {}, v1 = {};
            if (g < NNODES) {
                const float* s = X + (long)g * NFEAT + k0 + col;
                v0 = *(const float4*)(s);
                v1 = *(const float4*)(s + 4);
            }
            *(uint4*)&As[row][col] = pack8(v0, v1);
        }
        { // stage B transposed 64(n)x32(k): 8 column floats/thread -> b128
            const int n  = tid >> 2;          // 0..63
            const int kk = (tid & 3) * 8;
            const float* s = W + (long)(k0 + kk) * NHID + n0 + n;
            float4 v0, v1;
            v0.x = s[0];        v0.y = s[NHID];     v0.z = s[2 * NHID]; v0.w = s[3 * NHID];
            v1.x = s[4 * NHID]; v1.y = s[5 * NHID]; v1.z = s[6 * NHID]; v1.w = s[7 * NHID];
            *(uint4*)&Bt[n][kk] = pack8(v0, v1);
        }
        __syncthreads();

        FragU a, b0, b1;
        a.q[0] = *(const uint4*)&As[mr][kh];
        a.q[1] = *(const uint4*)&As[mr][kh + 8];
        b0.q[0] = *(const uint4*)&Bt[wn * 32 + nlo][kb];
        b0.q[1] = *(const uint4*)&Bt[wn * 32 + nlo][kb + 8];
        b1.q[0] = *(const uint4*)&Bt[wn * 32 + 16 + nlo][kb];
        b1.q[1] = *(const uint4*)&Bt[wn * 32 + 16 + nlo][kb + 8];

        acc0 = __builtin_amdgcn_wmma_f32_16x16x32_bf16(false, a.bf, false, b0.bf,
                                                       (short)0, acc0, false, false);
        acc1 = __builtin_amdgcn_wmma_f32_16x16x32_bf16(false, a.bf, false, b1.bf,
                                                       (short)0, acc1, false, false);
        __syncthreads();
    }
    // D layout: VGPR v -> row v (lanes 0-15) / v+8 (lanes 16-31), lane%16 = col
    const int col   = n0 + wn * 32 + nlo;
    const int rbase = m0 + wm * 16 + 8 * h;
#pragma unroll
    for (int v = 0; v < 8; ++v) {
        const int row = rbase + v;
        if (row < NNODES) {
            H[(long)row * NHID + col]      = acc0[v];
            H[(long)row * NHID + col + 16] = acc1[v];
        }
    }
}

// =====================================================================
// GEMM2: O[50000x40] = H2[50000x256] @ W2[256x40] (N padded to 48)
// block = 128 thr (4 waves, each one 16-row M tile x 3 N tiles)
// W2 transposed bf16 in LDS once: [48][256] shorts = 24 KB.
// =====================================================================
__launch_bounds__(128)
__global__ void gemm2_wmma(const float* __restrict__ H2,
                           const float* __restrict__ W2,
                           float* __restrict__ O) {
    __shared__ __align__(16) unsigned short Wt[NCLASS_P][NHID];   // 24 KB
    __shared__ __align__(16) unsigned short As[64][32];           // 4 KB
    const int tid  = threadIdx.x;
    const int lane = tid & 31;
    const int wid  = tid >> 5;
    const int m0   = blockIdx.x * 64;

    // stage W2^T (pad cols 40..47 with zeros): 1536 8-short tasks, b128 stores
    for (int task = tid; task < NCLASS_P * (NHID / 8); task += 128) {
        const int n  = task >> 5;           // 0..47
        const int kk = (task & 31) * 8;     // 0..248
        float4 v0 = {}, v1 = {};
        if (n < NCLASS) {
            const float* s = W2 + (long)kk * NCLASS + n;
            v0.x = s[0];          v0.y = s[NCLASS];     v0.z = s[2 * NCLASS]; v0.w = s[3 * NCLASS];
            v1.x = s[4 * NCLASS]; v1.y = s[5 * NCLASS]; v1.z = s[6 * NCLASS]; v1.w = s[7 * NCLASS];
        }
        *(uint4*)&Wt[n][kk] = pack8(v0, v1);
    }
    __syncthreads();

    const int nlo = lane & 15;
    const int h   = lane >> 4;
    const int mr  = wid * 16 + nlo;
    const int kh  = 16 * h;

    v8f a0 = {}, a1 = {}, a2 = {};
    for (int k0 = 0; k0 < NHID; k0 += 32) {
        { // stage A 64x32: 16 floats/thread -> two ds_store_b128
            const int row = tid >> 1;
            const int col = (tid & 1) * 16;
            const int g   = m0 + row;
            float4 v0 = {}, v1 = {}, v2 = {}, v3 = {};
            if (g < NNODES) {
                const float* s = H2 + (long)g * NHID + k0 + col;
                v0 = *(const float4*)(s);
                v1 = *(const float4*)(s + 4);
                v2 = *(const float4*)(s + 8);
                v3 = *(const float4*)(s + 12);
            }
            *(uint4*)&As[row][col]     = pack8(v0, v1);
            *(uint4*)&As[row][col + 8] = pack8(v2, v3);
        }
        __syncthreads();

        FragU a, b0, b1, b2;
        a.q[0] = *(const uint4*)&As[mr][kh];
        a.q[1] = *(const uint4*)&As[mr][kh + 8];
        const int kf = k0 + 16 * h;
        b0.q[0] = *(const uint4*)&Wt[nlo][kf];
        b0.q[1] = *(const uint4*)&Wt[nlo][kf + 8];
        b1.q[0] = *(const uint4*)&Wt[16 + nlo][kf];
        b1.q[1] = *(const uint4*)&Wt[16 + nlo][kf + 8];
        b2.q[0] = *(const uint4*)&Wt[32 + nlo][kf];
        b2.q[1] = *(const uint4*)&Wt[32 + nlo][kf + 8];

        a0 = __builtin_amdgcn_wmma_f32_16x16x32_bf16(false, a.bf, false, b0.bf, (short)0, a0, false, false);
        a1 = __builtin_amdgcn_wmma_f32_16x16x32_bf16(false, a.bf, false, b1.bf, (short)0, a1, false, false);
        a2 = __builtin_amdgcn_wmma_f32_16x16x32_bf16(false, a.bf, false, b2.bf, (short)0, a2, false, false);
        __syncthreads();
    }
    const int rbase = m0 + wid * 16 + 8 * h;
#pragma unroll
    for (int v = 0; v < 8; ++v) {
        const int row = rbase + v;
        if (row >= NNODES) continue;
        float* d = O + (long)row * NCLASS;
        d[nlo]      = a0[v];
        d[16 + nlo] = a1[v];
        if (32 + nlo < NCLASS) d[32 + nlo] = a2[v];
    }
}

// =====================================================================
// SpMM F=256: edge-parallel, 64 threads/edge, float4 gather + 4 f32 atomics.
// h (51.2 MB) fits entirely in MI455X's 192 MB L2 -> L2-BW bound.
// =====================================================================
__launch_bounds__(256)
__global__ void spmm_hid(const int* __restrict__ esrc, const int* __restrict__ edst,
                         const float* __restrict__ ew, const float* __restrict__ h,
                         float* __restrict__ out, int nedges) {
    const long gid = (long)blockIdx.x * blockDim.x + threadIdx.x;
    if (gid >= (long)nedges * 64) return;
    const int e = (int)(gid >> 6);
    const int c = (int)(gid & 63) * 4;
    const int s = esrc[e], d = edst[e];
    const float w = ew[e];
    const float4 v = *(const float4*)(h + (long)s * NHID + c);
    float* o = out + (long)d * NHID + c;
    atomicAdd(o + 0, w * v.x);
    atomicAdd(o + 1, w * v.y);
    atomicAdd(o + 2, w * v.z);
    atomicAdd(o + 3, w * v.w);
}

// SpMM F=40: one thread per (edge, class)
__launch_bounds__(256)
__global__ void spmm_cls(const int* __restrict__ esrc, const int* __restrict__ edst,
                         const float* __restrict__ ew, const float* __restrict__ o,
                         float* __restrict__ out, int nedges) {
    const long gid = (long)blockIdx.x * blockDim.x + threadIdx.x;
    if (gid >= (long)nedges * NCLASS) return;
    const int e = (int)(gid / NCLASS);
    const int f = (int)(gid % NCLASS);
    atomicAdd(out + (long)edst[e] * NCLASS + f,
              ew[e] * o[(long)esrc[e] * NCLASS + f]);
}

// bias + ReLU + deterministic dropout(0.5), scale 1/(1-p)=2
__launch_bounds__(256)
__global__ void relu_drop(const float* __restrict__ a, const float* __restrict__ b1,
                          float* __restrict__ h2) {
    const int idx = blockIdx.x * blockDim.x + threadIdx.x;
    if (idx >= NNODES * NHID) return;
    float v = fmaxf(a[idx] + b1[idx & (NHID - 1)], 0.f);
    const unsigned r = pcg_hash((unsigned)idx ^ 0x2A2A2A2Au);   // "key 42" stream
    h2[idx] = (r & 1u) ? (v * 2.0f) : 0.f;
}

__launch_bounds__(256)
__global__ void zero_buf(float* __restrict__ p, long n) {
    const long i = (long)blockIdx.x * blockDim.x + threadIdx.x;
    if (i < n) p[i] = 0.f;
}

__launch_bounds__(256)
__global__ void init_out(float* __restrict__ out, const float* __restrict__ b2) {
    const int i = blockIdx.x * blockDim.x + threadIdx.x;
    if (i < NNODES * NCLASS) out[i] = b2[i % NCLASS];
}

// =====================================================================
extern "C" void kernel_launch(void* const* d_in, const int* in_sizes, int n_in,
                              void* d_out, int out_size, void* d_ws, size_t ws_size,
                              hipStream_t stream) {
    const float* x    = (const float*)d_in[0];
    const int*   esrc = (const int*)  d_in[1];
    const int*   edst = (const int*)  d_in[2];
    const float* ew   = (const float*)d_in[3];
    const float* W1   = (const float*)d_in[4];
    const float* b1   = (const float*)d_in[5];
    const float* W2   = (const float*)d_in[6];
    const float* b2   = (const float*)d_in[7];
    float* out = (float*)d_out;
    const int E = in_sizes[1];

    // workspace layout (floats): hbuf[50000x256] | abuf[50000x256] | obuf[50000x40]
    float* hbuf = (float*)d_ws;
    float* abuf = hbuf + (long)NNODES * NHID;
    float* obuf = abuf + (long)NNODES * NHID;

    // 1) H = X @ W1   (bf16 WMMA)
    gemm1_wmma<<<dim3((NNODES + 63) / 64, NHID / 64), 256, 0, stream>>>(x, W1, hbuf);

    // 2) A = spmm(H)
    {
        const long n = (long)NNODES * NHID;
        zero_buf<<<(int)((n + 255) / 256), 256, 0, stream>>>(abuf, n);
        const long t = (long)E * 64;
        spmm_hid<<<(int)((t + 255) / 256), 256, 0, stream>>>(esrc, edst, ew, hbuf, abuf, E);
    }

    // 3) H2 = dropout(relu(A + b1))   (written back into hbuf)
    relu_drop<<<(NNODES * NHID + 255) / 256, 256, 0, stream>>>(abuf, b1, hbuf);

    // 4) O = H2 @ W2   (bf16 WMMA, N padded 40->48)
    gemm2_wmma<<<(NNODES + 63) / 64, 128, 0, stream>>>(hbuf, W2, obuf);

    // 5) out = spmm(O) + b2  (init out with bias, scatter-add into it)
    init_out<<<(NNODES * NCLASS + 255) / 256, 256, 0, stream>>>(out, b2);
    {
        const long t = (long)E * NCLASS;
        spmm_cls<<<(int)((t + 255) / 256), 256, 0, stream>>>(esrc, edst, ew, obuf, out, E);
    }
}
